// HippocampalShortcut_73126113182067
// MI455X (gfx1250) — matrix operand order repaired and checked by reading, not compiled, not run on hardware
//
#include <hip/hip_runtime.h>

using u16 = unsigned short;
using u32 = unsigned int;
using u64 = unsigned long long;

typedef float  v8f   __attribute__((ext_vector_type(8)));
typedef __bf16 v16bf __attribute__((ext_vector_type(16)));

struct FragBits { uint4 lo, hi; };   // 32 bytes = 16 bf16

// ---------------------------------------------------------------------------
// CDNA5 async global->LDS copy (ASYNCcnt-tracked, CDNA5 ISA)
// VDST VGPR = wave-relative LDS byte offset (low 32 bits of generic shared
// pointer, per the LDS aperture mapping), VADDR = 64-bit global address.
// ---------------------------------------------------------------------------
__device__ __forceinline__ void async_copy_b128(u32 ldsOff, u64 gaddr) {
    asm volatile("global_load_async_to_lds_b128 %0, %1, off"
                 :: "v"(ldsOff), "v"(gaddr)
                 : "memory");
}
__device__ __forceinline__ void wait_async_all() {
    asm volatile("s_wait_asynccnt 0x0" ::: "memory");
}

// ---------------------------------------------------------------------------
// LDS fragment loaders following CDNA5 WMMA VGPR layouts (CDNA5 ISA)
// A 16x32 bf16: lanes 0-15 hold M=lane, K = {0..7, 16..23}; lanes 16-31 hold
// same M with K = {8..15, 24..31}.
// B 32x16 bf16 (stored as Bw[N,K] rows): lane holds column n=lane&15, lanes
// 0-15 K = {0..7, 16..23}, lanes 16-31 K = {8..15, 24..31}.
// LDS tiles use a padded row stride of 40 halfwords (80 B) to spread banks.
// ---------------------------------------------------------------------------
__device__ __forceinline__ v16bf load_fragA(const u16* sA, int rowBase, int lane) {
    const int r  = rowBase + (lane & 15);
    const int ko = (lane >> 4) * 8;
    FragBits f;
    f.lo = *(const uint4*)(sA + r * 40 + ko);        // K = ko .. ko+7
    f.hi = *(const uint4*)(sA + r * 40 + ko + 16);   // K = ko+16 .. ko+23
    return __builtin_bit_cast(v16bf, f);
}

__device__ __forceinline__ v16bf load_fragB(const u16* sB, int rowBase, int lane) {
    const int r  = rowBase + (lane & 15);
    const int ko = (lane >> 4) * 8;
    FragBits f;
    f.lo = *(const uint4*)(sB + r * 40 + ko);        // K = ko .. ko+7
    f.hi = *(const uint4*)(sB + r * 40 + ko + 16);   // K = ko+16 .. ko+23
    return __builtin_bit_cast(v16bf, f);
}

// ---------------------------------------------------------------------------
// NT GEMM: C[M,N] = A[M,K] (row major, bf16) x Bw[N,K] (row major, bf16)^T
// fp32 accumulate via v_wmma_f32_16x16x32_bf16. Block tile 128x128, BK=32,
// double-buffered LDS fed by global_load_async_to_lds_b128. 256 threads =
// 8 waves (wave32); each wave owns a 32(M) x 64(N) sub-tile = 2x4 grid of
// 16x16 WMMA tiles -> 8 WMMAs per k-step per wave.
// ---------------------------------------------------------------------------
#define TILE_HW (128 * 40)          // halfwords per LDS tile (padded)
#define TILE_BYTES (TILE_HW * 2)    // 10240 B

__global__ __launch_bounds__(256)
void gemm_nt_bf16(const u16* __restrict__ A, const u16* __restrict__ Bw,
                  float* __restrict__ C, int M, int N, int K, int doRelu)
{
    __shared__ __align__(16) u16 sA[2][TILE_HW];
    __shared__ __align__(16) u16 sB[2][TILE_HW];

    const int tid   = threadIdx.x;
    const int lane  = tid & 31;
    const int wave  = tid >> 5;
    const int waveM = wave & 3;     // 4 waves along M
    const int waveN = wave >> 2;    // 2 waves along N
    const long mTile = (long)blockIdx.y * 128;
    const long nTile = (long)blockIdx.x * 128;

    // Global->LDS staging: 128 rows x 32 cols bf16 per tile = 512 chunks of
    // 16 B; each thread owns chunks tid and tid+256.
    const int c0 = tid, c1 = tid + 256;
    const int r0 = c0 >> 2, col0 = (c0 & 3) * 8;
    const int r1 = c1 >> 2, col1 = (c1 & 3) * 8;

    const u16* gA0 = A  + (mTile + r0) * (long)K + col0;
    const u16* gA1 = A  + (mTile + r1) * (long)K + col1;
    const u16* gB0 = Bw + (nTile + r0) * (long)K + col0;
    const u16* gB1 = Bw + (nTile + r1) * (long)K + col1;

    // per-thread LDS byte offsets for this thread's two 16B chunks (buffer 0)
    const u32 aOff0 = (u32)(size_t)(&sA[0][r0 * 40 + col0]);
    const u32 aOff1 = (u32)(size_t)(&sA[0][r1 * 40 + col1]);
    const u32 bOff0 = (u32)(size_t)(&sB[0][r0 * 40 + col0]);
    const u32 bOff1 = (u32)(size_t)(&sB[0][r1 * 40 + col1]);

    v8f acc[2][4] = {};

    const int nk = K >> 5;

    // prologue: tile 0 -> buffer 0
    async_copy_b128(aOff0, (u64)(gA0));
    async_copy_b128(aOff1, (u64)(gA1));
    async_copy_b128(bOff0, (u64)(gB0));
    async_copy_b128(bOff1, (u64)(gB1));

    for (int kt = 0; kt < nk; ++kt) {
        wait_async_all();       // this wave's tile-kt loads landed in LDS
        __syncthreads();        // all waves' writes visible; prev buffer free

        if (kt + 1 < nk) {      // overlap tile kt+1 loads with compute of kt
            const long ko = (long)(kt + 1) << 5;
            const u32 bo = (u32)(((kt + 1) & 1) * TILE_BYTES);
            async_copy_b128(aOff0 + bo, (u64)(gA0 + ko));
            async_copy_b128(aOff1 + bo, (u64)(gA1 + ko));
            async_copy_b128(bOff0 + bo, (u64)(gB0 + ko));
            async_copy_b128(bOff1 + bo, (u64)(gB1 + ko));
        }

        const u16* tA = sA[kt & 1];
        const u16* tB = sB[kt & 1];
        const v16bf a0 = load_fragA(tA, waveM * 32,      lane);
        const v16bf a1 = load_fragA(tA, waveM * 32 + 16, lane);
        const v16bf b0 = load_fragB(tB, waveN * 64,      lane);
        const v16bf b1 = load_fragB(tB, waveN * 64 + 16, lane);
        const v16bf b2 = load_fragB(tB, waveN * 64 + 32, lane);
        const v16bf b3 = load_fragB(tB, waveN * 64 + 48, lane);

        acc[0][0] = __builtin_amdgcn_wmma_f32_16x16x32_bf16(false, a0, false, b0, (short)0, acc[0][0], false, false);
        acc[0][1] = __builtin_amdgcn_wmma_f32_16x16x32_bf16(false, a0, false, b1, (short)0, acc[0][1], false, false);
        acc[0][2] = __builtin_amdgcn_wmma_f32_16x16x32_bf16(false, a0, false, b2, (short)0, acc[0][2], false, false);
        acc[0][3] = __builtin_amdgcn_wmma_f32_16x16x32_bf16(false, a0, false, b3, (short)0, acc[0][3], false, false);
        acc[1][0] = __builtin_amdgcn_wmma_f32_16x16x32_bf16(false, a1, false, b0, (short)0, acc[1][0], false, false);
        acc[1][1] = __builtin_amdgcn_wmma_f32_16x16x32_bf16(false, a1, false, b1, (short)0, acc[1][1], false, false);
        acc[1][2] = __builtin_amdgcn_wmma_f32_16x16x32_bf16(false, a1, false, b2, (short)0, acc[1][2], false, false);
        acc[1][3] = __builtin_amdgcn_wmma_f32_16x16x32_bf16(false, a1, false, b3, (short)0, acc[1][3], false, false);
    }

    // Epilogue per C/D layout: lane holds column n = lane&15; VGPR r holds
    // row r (lanes 0-15) or 8+r (lanes 16-31).
    #pragma unroll
    for (int i = 0; i < 2; ++i) {
        const long mBase = mTile + waveM * 32 + i * 16 + (lane >> 4) * 8;
        #pragma unroll
        for (int j = 0; j < 4; ++j) {
            const long n = nTile + waveN * 64 + j * 16 + (lane & 15);
            #pragma unroll
            for (int r = 0; r < 8; ++r) {
                float v = acc[i][j][r];
                if (doRelu) v = fmaxf(v, 0.0f);
                C[(mBase + r) * N + n] = v;
            }
        }
    }
}

// ---------------------------------------------------------------------------
// kWTA (top-k threshold via 24-step binary search) + L2 normalize, emit bf16.
// One 256-thread block per row of 4096 fp32 values (16 per thread).
// ---------------------------------------------------------------------------
__global__ __launch_bounds__(256)
void kwta_l2_bf16(const float* __restrict__ H, u16* __restrict__ O,
                  int ncols, const int* __restrict__ kptr)
{
    const int row  = blockIdx.x;
    const int tid  = threadIdx.x;
    const int lane = tid & 31;
    const int wave = tid >> 5;
    const int k    = *kptr;

    const float* hr = H + (long)row * ncols;
    float v[16];
    #pragma unroll
    for (int i = 0; i < 16; ++i) v[i] = hr[tid + (i << 8)];

    __shared__ float sF[8];
    __shared__ int   sI[8];

    // row max (values are >= 0 post-relu)
    float mx = 0.0f;
    #pragma unroll
    for (int i = 0; i < 16; ++i) mx = fmaxf(mx, v[i]);
    #pragma unroll
    for (int o = 16; o > 0; o >>= 1) mx = fmaxf(mx, __shfl_xor(mx, o, 32));
    if (lane == 0) sF[wave] = mx;
    __syncthreads();
    mx = sF[0];
    #pragma unroll
    for (int w = 1; w < 8; ++w) mx = fmaxf(mx, sF[w]);
    __syncthreads();

    float lo = 0.0f, hi = mx;
    for (int it = 0; it < 24; ++it) {
        const float mid = 0.5f * (lo + hi);
        int c = 0;
        #pragma unroll
        for (int i = 0; i < 16; ++i) c += (v[i] > mid) ? 1 : 0;
        #pragma unroll
        for (int o = 16; o > 0; o >>= 1) c += __shfl_xor(c, o, 32);
        if (lane == 0) sI[wave] = c;
        __syncthreads();
        int tot = 0;
        #pragma unroll
        for (int w = 0; w < 8; ++w) tot += sI[w];
        __syncthreads();
        if (tot >= k) lo = mid; else hi = mid;
    }
    const float thr = lo;

    float ss = 0.0f;
    #pragma unroll
    for (int i = 0; i < 16; ++i) if (v[i] > thr) ss += v[i] * v[i];
    #pragma unroll
    for (int o = 16; o > 0; o >>= 1) ss += __shfl_xor(ss, o, 32);
    if (lane == 0) sF[wave] = ss;
    __syncthreads();
    ss = 0.0f;
    #pragma unroll
    for (int w = 0; w < 8; ++w) ss += sF[w];
    const float inv = rsqrtf(fmaxf(ss, 1e-20f));

    u16* orow = O + (long)row * ncols;
    #pragma unroll
    for (int i = 0; i < 16; ++i) {
        const float o = (v[i] > thr) ? v[i] * inv : 0.0f;
        const u32 u = __float_as_uint(o);
        orow[tid + (i << 8)] = (u16)((u + 0x7FFFu + ((u >> 16) & 1u)) >> 16);  // RNE
    }
}

// ---------------------------------------------------------------------------
// fp32 -> bf16 (round to nearest even)
// ---------------------------------------------------------------------------
__global__ void f32_to_bf16(const float* __restrict__ in, u16* __restrict__ out, long n)
{
    const long i = (long)blockIdx.x * blockDim.x + threadIdx.x;
    if (i < n) {
        const u32 u = __float_as_uint(in[i]);
        out[i] = (u16)((u + 0x7FFFu + ((u >> 16) & 1u)) >> 16);
    }
}

// ---------------------------------------------------------------------------
// out = x + sigmoid(glog + bg) * pred
// ---------------------------------------------------------------------------
__global__ void gate_residual(const float* __restrict__ x, const float* __restrict__ pred,
                              const float* __restrict__ glog, const float* __restrict__ bg,
                              float* __restrict__ out, long total, int dmask)
{
    const long i = (long)blockIdx.x * blockDim.x + threadIdx.x;
    if (i < total) {
        const float z = glog[i] + bg[(int)(i & (long)dmask)];
        const float g = 1.0f / (1.0f + __expf(-z));
        out[i] = x[i] + g * pred[i];
    }
}

// ---------------------------------------------------------------------------
extern "C" void kernel_launch(void* const* d_in, const int* in_sizes, int n_in,
                              void* d_out, int out_size, void* d_ws, size_t ws_size,
                              hipStream_t stream)
{
    (void)in_sizes; (void)n_in; (void)out_size; (void)ws_size;

    const float* x   = (const float*)d_in[0];   // [4,2048,1024]
    const float* Wd  = (const float*)d_in[1];   // [4096,1024]
    const float* W   = (const float*)d_in[2];   // [4096,4096]
    const float* Wu  = (const float*)d_in[3];   // [1024,4096]
    const float* Wg  = (const float*)d_in[4];   // [1024,1024]
    const float* bg  = (const float*)d_in[5];   // [1024]
    const int*   kp  = (const int*)d_in[6];     // scalar k

    const long Bsz = 4, T = 2048, D = 1024, N = 4096;
    const long BT = Bsz * T;                    // 8192

    // bump-allocate scratch from d_ws (h2 aliases h, h2_bf16 aliases h_bf16)
    char* ws = (char*)d_ws;
    size_t off = 0;
    auto take = [&](size_t bytes) -> void* {
        off = (off + 255) & ~(size_t)255;
        void* p = ws + off;
        off += bytes;
        return p;
    };
    u16*   xb  = (u16*)  take((size_t)BT * D * 2);
    u16*   Wdb = (u16*)  take((size_t)N  * D * 2);
    u16*   Wb  = (u16*)  take((size_t)N  * N * 2);
    u16*   Wub = (u16*)  take((size_t)D  * N * 2);
    u16*   Wgb = (u16*)  take((size_t)D  * D * 2);
    float* hf  = (float*)take((size_t)BT * N * 4);  // h, then reused for h2
    u16*   hb  = (u16*)  take((size_t)BT * N * 2);  // h_bf16, then h2_bf16
    float* pr  = (float*)take((size_t)BT * D * 4);  // prediction
    float* gl  = (float*)take((size_t)BT * D * 4);  // gate logits

    auto cvt = [&](const float* src, u16* dst, long n) {
        f32_to_bf16<<<dim3((unsigned)((n + 255) / 256)), dim3(256), 0, stream>>>(src, dst, n);
    };
    cvt(x,  xb,  BT * D);
    cvt(Wd, Wdb, N * D);
    cvt(W,  Wb,  N * N);
    cvt(Wu, Wub, D * N);
    cvt(Wg, Wgb, D * D);

    // h = relu(x @ Wd^T)            [BT x N], K = D
    gemm_nt_bf16<<<dim3((unsigned)(N / 128), (unsigned)(BT / 128)), dim3(256), 0, stream>>>(
        xb, Wdb, hf, (int)BT, (int)N, (int)D, 1);
    kwta_l2_bf16<<<dim3((unsigned)BT), dim3(256), 0, stream>>>(hf, hb, (int)N, kp);

    // h2 = relu(h @ W^T)            [BT x N], K = N  (reuses hf)
    gemm_nt_bf16<<<dim3((unsigned)(N / 128), (unsigned)(BT / 128)), dim3(256), 0, stream>>>(
        hb, Wb, hf, (int)BT, (int)N, (int)N, 1);
    kwta_l2_bf16<<<dim3((unsigned)BT), dim3(256), 0, stream>>>(hf, hb, (int)N, kp);

    // prediction = h2 @ Wu^T        [BT x D], K = N
    gemm_nt_bf16<<<dim3((unsigned)(D / 128), (unsigned)(BT / 128)), dim3(256), 0, stream>>>(
        hb, Wub, pr, (int)BT, (int)D, (int)N, 0);

    // gate logits = x @ Wg^T        [BT x D], K = D
    gemm_nt_bf16<<<dim3((unsigned)(D / 128), (unsigned)(BT / 128)), dim3(256), 0, stream>>>(
        xb, Wgb, gl, (int)BT, (int)D, (int)D, 0);

    // out = x + sigmoid(glog + bg) * pred
    const long total = BT * D;
    gate_residual<<<dim3((unsigned)((total + 255) / 256)), dim3(256), 0, stream>>>(
        x, pr, gl, bg, (float*)d_out, total, (int)(D - 1));
}
